// GaussianSplatting_13597866459687
// MI455X (gfx1250) — matrix operand order repaired and checked by reading, not compiled, not run on hardware
//
#include <hip/hip_runtime.h>

// ---------------------------------------------------------------------------
// GaussianSplatting as GEMM:  canvas = (gy*act)^T @ gx,  K = 32768
//   Phase 1: materialize gxT[w][n], gyaT[h][n] as f16 (K-contiguous, 32MB each)
//   Phase 2: WMMA f32_16x16x32_f16 GEMM, 16-way K-split into f32 partials
//   Phase 3: deterministic reduction of partials -> canvas (f32)
// Workspace: 32MB (gxT) + 32MB (gyaT) + 16MB (partials) = 80MB  (< 192MB L2)
// ---------------------------------------------------------------------------

#define RES     512
#define NPTS    32768
#define SN      32768                 // K stride (halfs) of transposed mats
#define NSPLIT  16
#define KSPLIT  (NPTS / NSPLIT)       // 2048
#define SIGMA_F 0.02f

typedef __attribute__((ext_vector_type(16))) _Float16 v16h;
typedef __attribute__((ext_vector_type(8)))  _Float16 v8h;
typedef __attribute__((ext_vector_type(8)))  float    v8f;

// ---------------- Phase 1: Gaussian factor matrices (f16, transposed) -------
// blockIdx.x in [0,1024): which = idx>>9 (0 -> gxT, 1 -> gyaT), row = idx&511
__global__ __launch_bounds__(256)
void fill_factors(const float* __restrict__ pos,      // [NPTS][2]
                  const float* __restrict__ act,      // [NPTS]
                  const float* __restrict__ scale_p,  // [1]
                  _Float16* __restrict__ gxT,         // [RES][SN]
                  _Float16* __restrict__ gyaT)        // [RES][SN]
{
    const float s       = scale_p[0];
    const float inv2s2  = 1.0f / (2.0f * SIGMA_F * SIGMA_F);
    const int   row     = blockIdx.x & (RES - 1);
    const int   which   = blockIdx.x >> 9;
    const float g       = (float)row * (1.0f / (float)(RES - 1));

    if (which == 0) {
        _Float16* __restrict__ dst = gxT + (size_t)row * SN;
        for (int n = threadIdx.x; n < NPTS; n += 256) {
            float px = pos[2 * n + 0];
            float x  = 0.5f + s * (px - 0.5f);
            float d  = g - x;
            dst[n]   = (_Float16)__expf(-d * d * inv2s2);
        }
    } else {
        _Float16* __restrict__ dst = gyaT + (size_t)row * SN;
        for (int n = threadIdx.x; n < NPTS; n += 256) {
            float py = pos[2 * n + 1];
            float y  = 1.0f - (0.5f + s * (py - 0.5f));
            float d  = g - y;
            dst[n]   = (_Float16)(__expf(-d * d * inv2s2) * act[n]);
        }
    }
}

// ---------------- Phase 2: WMMA GEMM with K-split ---------------------------
// 1024 waves total: wave = split*64 + blk; blk -> 64x64 output macro-tile.
// Each wave: 4x4 tiles of 16x16, K loop over KSPLIT in steps of 32.
__global__ __launch_bounds__(256)
void gemm_wmma(const _Float16* __restrict__ gyaT,   // A^T storage: [RES][SN]
               const _Float16* __restrict__ gxT,    // B^T storage: [RES][SN]
               float* __restrict__ partial)         // [NSPLIT][RES][RES]
{
    const int wave  = (int)((blockIdx.x * 256u + threadIdx.x) >> 5); // 0..1023
    const int lane  = (int)(threadIdx.x & 31u);
    const int split = wave >> 6;          // 0..15
    const int blk   = wave & 63;          // 0..63
    const int h0    = (blk >> 3) << 6;    // 64-aligned output row
    const int w0    = (blk & 7) << 6;     // 64-aligned output col
    const int m     = lane & 15;          // row/col within 16
    const int grp   = lane >> 4;          // half-wave group

    const int kbeg = split * KSPLIT;
    const int kend = kbeg + KSPLIT;

    v8f acc[4][4];
    const v8f vzero = {};
#pragma unroll
    for (int i = 0; i < 4; ++i)
#pragma unroll
        for (int j = 0; j < 4; ++j) acc[i][j] = vzero;

    // Per-lane base pointers (K-contiguous rows).
    // A frag (16x32 f16): elems 0-15 = K[grp*16 .. +15]
    // B frag (32x16 f16): elems 0-15 = K[grp*16 .. +15]
    const _Float16* aBase[4];
    const _Float16* bBase[4];
#pragma unroll
    for (int t = 0; t < 4; ++t) {
        aBase[t] = gyaT + (size_t)(h0 + t * 16 + m) * SN + grp * 16;
        bBase[t] = gxT  + (size_t)(w0 + t * 16 + m) * SN + grp * 16;
    }

    for (int k0 = kbeg; k0 < kend; k0 += 32) {
        v16h a[4], b[4];
#pragma unroll
        for (int t = 0; t < 4; ++t) {
            v8h lo = *(const v8h*)(aBase[t] + k0);
            v8h hi = *(const v8h*)(aBase[t] + k0 + 8);
            a[t] = __builtin_shufflevector(lo, hi,
                       0, 1, 2, 3, 4, 5, 6, 7, 8, 9, 10, 11, 12, 13, 14, 15);
            b[t] = *(const v16h*)(bBase[t] + k0);
        }
#pragma unroll
        for (int i = 0; i < 4; ++i)
#pragma unroll
            for (int j = 0; j < 4; ++j)
                acc[i][j] = __builtin_amdgcn_wmma_f32_16x16x32_f16(
                    /*neg_a=*/false, a[i], /*neg_b=*/false, b[j],
                    /*c_mod=*/(short)0, acc[i][j],
                    /*reuse_a=*/false, /*reuse_b=*/false);
    }

    // C/D layout: VGPR v -> row (v + 8*grp), col = m  (within each 16x16 tile)
    float* __restrict__ p = partial + (size_t)split * RES * RES;
#pragma unroll
    for (int i = 0; i < 4; ++i) {
#pragma unroll
        for (int j = 0; j < 4; ++j) {
            const int hh = h0 + i * 16 + 8 * grp;
            const int ww = w0 + j * 16 + m;
#pragma unroll
            for (int v = 0; v < 8; ++v)
                p[(size_t)(hh + v) * RES + ww] = acc[i][j][v];
        }
    }
}

// ---------------- Phase 3: deterministic reduction --------------------------
__global__ __launch_bounds__(256)
void reduce_partials(const float* __restrict__ partial,  // [NSPLIT][RES*RES]
                     float* __restrict__ out)            // [RES*RES]
{
    const int i = (int)(blockIdx.x * 256u + threadIdx.x);
    float s = 0.0f;
#pragma unroll
    for (int sp = 0; sp < NSPLIT; ++sp)
        s += partial[(size_t)sp * (RES * RES) + i];
    out[i] = s;
}

// ---------------------------------------------------------------------------
extern "C" void kernel_launch(void* const* d_in, const int* in_sizes, int n_in,
                              void* d_out, int out_size, void* d_ws, size_t ws_size,
                              hipStream_t stream) {
    const float* pos   = (const float*)d_in[0];  // [NPTS][2] f32
    const float* act   = (const float*)d_in[1];  // [NPTS]    f32
    const float* scale = (const float*)d_in[2];  // [1]       f32
    float*       out   = (float*)d_out;          // [RES][RES] f32

    char* ws = (char*)d_ws;
    _Float16* gxT     = (_Float16*)(ws);                                  // 32MB
    _Float16* gyaT    = (_Float16*)(ws + (size_t)RES * SN * 2);           // 32MB
    float*    partial = (float*)   (ws + (size_t)2 * RES * SN * 2);       // 16MB

    // Phase 1: 1024 rows (512 gxT + 512 gyaT), 256 threads each
    fill_factors<<<1024, 256, 0, stream>>>(pos, act, scale, gxT, gyaT);

    // Phase 2: 1024 waves = 128 blocks x 8 waves
    gemm_wmma<<<128, 256, 0, stream>>>(gyaT, gxT, partial);

    // Phase 3: RES*RES / 256 blocks
    reduce_partials<<<(RES * RES) / 256, 256, 0, stream>>>(partial, out);
}
